// TransformerLanguageModel_33260226740540
// MI455X (gfx1250) — matrix operand order repaired and checked by reading, not compiled, not run on hardware
//
#include <hip/hip_runtime.h>
#include <hip/hip_bf16.h>

// ---------------------------------------------------------------------------
// Transformer LM forward for MI455X (gfx1250, wave32, WMMA).
// All matmuls run through v_wmma_f32_16x16x32_bf16 (fp32 accum).
// GEMM: block = 256 thr (8 waves), block tile 128x64, K-step 64,
// wave tile 32x32 (4 accumulators, 8 WMMAs per k-step, fragments reused 2x).
// B operand is ALWAYS N x K ("B^T" form) -> fully vectorized LDS staging
// (global_load_b128 + ds_store_b128); weights are transposed during the
// fp32->bf16 conversion, V is written transposed by the V-projection GEMM.
// ---------------------------------------------------------------------------

typedef __attribute__((ext_vector_type(16))) __bf16 v16bf;
typedef __attribute__((ext_vector_type(8)))  float  v8f;

#define TLM_B   4
#define TLM_S   1024
#define TLM_D   1024
#define TLM_H   16
#define TLM_HD  64
#define TLM_FF  4096
#define TLM_L   8
#define TLM_V   32000
#define TLM_T   (TLM_B * TLM_S)   // 4096 tokens

__device__ __forceinline__ unsigned short f2bf(float f) {
    unsigned u = __float_as_uint(f);
    unsigned r = u + 0x7FFFu + ((u >> 16) & 1u);   // round-to-nearest-even
    return (unsigned short)(r >> 16);
}

// ---------------------------------------------------------------------------
// fp32 -> bf16 conversion WITH transpose:  W (K x N, f32) -> Wt (N x K, bf16)
// Tiled 32x32 through LDS; coalesced on both sides.
// ---------------------------------------------------------------------------
__global__ __launch_bounds__(256) void k_f32_to_bf16_t(const float* __restrict__ W,
                                                       unsigned short* __restrict__ Wt,
                                                       int K, int N) {
    __shared__ float tile[32][33];
    int n0 = blockIdx.x * 32;
    int k0 = blockIdx.y * 32;
    int tid = threadIdx.x;

    int tk  = tid >> 3;            // 0..31
    int tn4 = (tid & 7) << 2;      // 0,4,...,28
    float4 v = *reinterpret_cast<const float4*>(W + (long long)(k0 + tk) * N + n0 + tn4);
    tile[tk][tn4 + 0] = v.x; tile[tk][tn4 + 1] = v.y;
    tile[tk][tn4 + 2] = v.z; tile[tk][tn4 + 3] = v.w;
    __syncthreads();

    int tn  = tid >> 3;            // 0..31
    int tk4 = (tid & 7) << 2;      // 0,4,...,28
    unsigned short e0 = f2bf(tile[tk4 + 0][tn]);
    unsigned short e1 = f2bf(tile[tk4 + 1][tn]);
    unsigned short e2 = f2bf(tile[tk4 + 2][tn]);
    unsigned short e3 = f2bf(tile[tk4 + 3][tn]);
    uint2 o;
    o.x = (unsigned)e0 | ((unsigned)e1 << 16);
    o.y = (unsigned)e2 | ((unsigned)e3 << 16);
    *reinterpret_cast<uint2*>(Wt + (long long)(n0 + tn) * K + k0 + tk4) = o;
}

// ---------------------------------------------------------------------------
// Embedding + sinusoidal positional encoding
// ---------------------------------------------------------------------------
__global__ __launch_bounds__(256) void k_embed_pe(const int* __restrict__ tok,
                                                  const float* __restrict__ emb,
                                                  float* __restrict__ x) {
    int t   = blockIdx.x;           // token index 0..T-1
    int pos = t % TLM_S;
    int v   = tok[t];
    const float nlog = -9.210340371976184f / (float)TLM_D;  // -ln(10000)/D
    for (int d = threadIdx.x; d < TLM_D; d += 256) {
        int   i2   = d & ~1;
        float freq = __expf((float)i2 * nlog);
        float a    = (float)pos * freq;
        float pe   = (d & 1) ? __cosf(a) : __sinf(a);
        x[(long long)t * TLM_D + d] = emb[(long long)v * TLM_D + d] + pe;
    }
}

// ---------------------------------------------------------------------------
// LayerNorm over last dim (D), bf16 output (feeds WMMA GEMMs)
// ---------------------------------------------------------------------------
__global__ __launch_bounds__(256) void k_layernorm_bf16(const float* __restrict__ x,
                                                        const float* __restrict__ g,
                                                        const float* __restrict__ b,
                                                        unsigned short* __restrict__ out,
                                                        int Dd) {
    int row = blockIdx.x;
    const float* xr = x + (long long)row * Dd;
    __shared__ float red[256];
    int tid = threadIdx.x;

    float s = 0.f, s2 = 0.f;
    for (int d = tid; d < Dd; d += 256) { float v = xr[d]; s += v; s2 += v * v; }

    red[tid] = s; __syncthreads();
    for (int o = 128; o > 0; o >>= 1) { if (tid < o) red[tid] += red[tid + o]; __syncthreads(); }
    float mean = red[0] / (float)Dd;
    __syncthreads();

    red[tid] = s2; __syncthreads();
    for (int o = 128; o > 0; o >>= 1) { if (tid < o) red[tid] += red[tid + o]; __syncthreads(); }
    float var  = red[0] / (float)Dd - mean * mean;
    float rstd = rsqrtf(var + 1e-5f);

    for (int d = tid; d < Dd; d += 256) {
        float v = (xr[d] - mean) * rstd * g[d] + b[d];
        out[(long long)row * Dd + d] = f2bf(v);
    }
}

// ---------------------------------------------------------------------------
// Causal softmax over rows of (B,H,S,S) scores -> bf16 probabilities
// ---------------------------------------------------------------------------
__global__ __launch_bounds__(256) void k_softmax_causal(const float* __restrict__ sc,
                                                        unsigned short* __restrict__ pr,
                                                        int Ss) {
    long long rid = blockIdx.x;          // (b*H+h)*S + i
    int i = (int)(rid % Ss);
    const float* srow = sc + rid * (long long)Ss;
    unsigned short* prow = pr + rid * (long long)Ss;
    __shared__ float red[256];
    int tid = threadIdx.x;

    float mx = -3.0e38f;
    for (int j = tid; j <= i; j += 256) mx = fmaxf(mx, srow[j]);
    red[tid] = mx; __syncthreads();
    for (int o = 128; o > 0; o >>= 1) { if (tid < o) red[tid] = fmaxf(red[tid], red[tid + o]); __syncthreads(); }
    mx = red[0]; __syncthreads();

    float sum = 0.f;
    for (int j = tid; j <= i; j += 256) sum += __expf(srow[j] - mx);
    red[tid] = sum; __syncthreads();
    for (int o = 128; o > 0; o >>= 1) { if (tid < o) red[tid] += red[tid + o]; __syncthreads(); }
    sum = red[0];

    float inv = 1.0f / sum;
    for (int j = tid; j < Ss; j += 256) {
        float v = (j <= i) ? __expf(srow[j] - mx) * inv : 0.0f;
        prow[j] = f2bf(v);
    }
}

// ---------------------------------------------------------------------------
// Batched bf16 WMMA GEMM.   C = act( scale * (A x Bt^T) + bias ) + resid
//   A : M x K bf16 (row-major, lda)
//   Bt: N x K bf16 (row-major, ldb)        <- always "transposed" form
//   Batch z: offset = (z / batchInner)*s?0 + (z % batchInner)*s?1
// Block tile 128(M) x 64(N), K-step 64; 8 waves, wave tile 32x32:
//   4 accumulators, 8 v_wmma per k-step, each fragment reused twice
//   (2 ds_load_b128 per wmma).
// LDS: As[m][k] and Bs[n][k] both plain (k = 16*half + s in each fragment),
// so every fragment is 32 contiguous bytes -> ds_load_b128.
// OUT_BT stores bf16 transposed: outB[gn*ldct + gm]  (batch==1 call sites).
// Requires M%128==0, N%64==0, K%64==0 (true at every call site).
// ---------------------------------------------------------------------------
template <int HAS_BIAS, int RELU, int HAS_RES, int OUT_F, int OUT_B, int OUT_BT>
__global__ __launch_bounds__(256) void k_gemm_bf16_wmma(
    const unsigned short* __restrict__ A, const unsigned short* __restrict__ Bt,
    const float* __restrict__ bias, const float* __restrict__ resid,
    float* __restrict__ outF, unsigned short* __restrict__ outB,
    int M, int N, int K, int lda, int ldb, int ldc, int ldct,
    long long sA0, long long sA1, long long sB0, long long sB1,
    long long sC0, long long sC1,
    int batchInner, float scale)
{
    __shared__ unsigned short As[128][64];  // [m][k]
    __shared__ unsigned short Bs[64][64];   // [n][k]

    int z  = blockIdx.z;
    int bo = z / batchInner, bi = z % batchInner;
    A  += (long long)bo * sA0 + (long long)bi * sA1;
    Bt += (long long)bo * sB0 + (long long)bi * sB1;
    long long coff = (long long)bo * sC0 + (long long)bi * sC1;

    int i0 = blockIdx.y * 128;
    int j0 = blockIdx.x * 64;

    int tid  = threadIdx.x;
    int lane = tid & 31;
    int wave = tid >> 5;
    int wm   = wave >> 1;        // 0..3  (M sub-tile, 32 rows)
    int wn   = wave & 1;         // 0..1  (N sub-tile, 32 cols)
    int half = lane >> 4;        // lane group 0/1
    int l16  = lane & 15;

    // Staging: thread r = tid>>2 handles rows r and r+64 (A) / row r (B),
    // group ag = tid&3 covers 8 k, plus a second chunk at +32 k.
    // A dest k-groups kept in source order (same layout as B).
    int r  = tid >> 2;
    int ag = tid & 3;
    int ac = ag << 3;
    int apc = ac;

    const unsigned short* aPtr0 = A + (long long)(i0 + r) * lda + ac;
    const unsigned short* aPtr1 = A + (long long)(i0 + r + 64) * lda + ac;
    const unsigned short* bPtr  = Bt + (long long)(j0 + r) * ldb + ac;
    unsigned short* asD0 = &As[r][apc];
    unsigned short* asD1 = &As[r][32 + apc];
    unsigned short* asD2 = &As[r + 64][apc];
    unsigned short* asD3 = &As[r + 64][32 + apc];
    unsigned short* bsD0 = &Bs[r][ac];
    unsigned short* bsD1 = &Bs[r][32 + ac];

    v8f acc00 = {0.f,0.f,0.f,0.f,0.f,0.f,0.f,0.f};
    v8f acc01 = acc00, acc10 = acc00, acc11 = acc00;

    int m0 = wm * 32 + l16;      // first 16-row A frag
    int m1 = m0 + 16;            // second
    int n0 = wn * 32 + l16;      // first 16-col B frag
    int n1 = n0 + 16;            // second
    int fo = half << 4;          // fragment k-offset within 32-chunk

    for (int k0 = 0; k0 < K; k0 += 64) {
        uint4 a0 = *reinterpret_cast<const uint4*>(aPtr0);
        uint4 a1 = *reinterpret_cast<const uint4*>(aPtr0 + 32);
        uint4 a2 = *reinterpret_cast<const uint4*>(aPtr1);
        uint4 a3 = *reinterpret_cast<const uint4*>(aPtr1 + 32);
        uint4 b0 = *reinterpret_cast<const uint4*>(bPtr);
        uint4 b1 = *reinterpret_cast<const uint4*>(bPtr + 32);
        aPtr0 += 64; aPtr1 += 64; bPtr += 64;
        *reinterpret_cast<uint4*>(asD0) = a0;
        *reinterpret_cast<uint4*>(asD1) = a1;
        *reinterpret_cast<uint4*>(asD2) = a2;
        *reinterpret_cast<uint4*>(asD3) = a3;
        *reinterpret_cast<uint4*>(bsD0) = b0;
        *reinterpret_cast<uint4*>(bsD1) = b1;
        __syncthreads();

        #pragma unroll
        for (int c = 0; c < 2; ++c) {
            union Frag { v16bf v; uint4 q[2]; } af0, af1, bf0, bf1;
            int cb = c * 32 + fo;
            af0.q[0] = *reinterpret_cast<const uint4*>(&As[m0][cb]);
            af0.q[1] = *reinterpret_cast<const uint4*>(&As[m0][cb + 8]);
            af1.q[0] = *reinterpret_cast<const uint4*>(&As[m1][cb]);
            af1.q[1] = *reinterpret_cast<const uint4*>(&As[m1][cb + 8]);
            bf0.q[0] = *reinterpret_cast<const uint4*>(&Bs[n0][cb]);
            bf0.q[1] = *reinterpret_cast<const uint4*>(&Bs[n0][cb + 8]);
            bf1.q[0] = *reinterpret_cast<const uint4*>(&Bs[n1][cb]);
            bf1.q[1] = *reinterpret_cast<const uint4*>(&Bs[n1][cb + 8]);

            acc00 = __builtin_amdgcn_wmma_f32_16x16x32_bf16(
                false, af0.v, false, bf0.v, (short)0, acc00, false, false);
            acc01 = __builtin_amdgcn_wmma_f32_16x16x32_bf16(
                false, af0.v, false, bf1.v, (short)0, acc01, false, false);
            acc10 = __builtin_amdgcn_wmma_f32_16x16x32_bf16(
                false, af1.v, false, bf0.v, (short)0, acc10, false, false);
            acc11 = __builtin_amdgcn_wmma_f32_16x16x32_bf16(
                false, af1.v, false, bf1.v, (short)0, acc11, false, false);
        }
        __syncthreads();
    }

    // ---- epilogue: C/D layout -> vgpr rr, lane: M = rr + 8*half, N = lane%16
    union Cvt { v8f v; float f[8]; };
    Cvt cf[2][2];
    cf[0][0].v = acc00; cf[0][1].v = acc01;
    cf[1][0].v = acc10; cf[1][1].v = acc11;
    #pragma unroll
    for (int u = 0; u < 2; ++u) {          // M sub-frag
        #pragma unroll
        for (int t = 0; t < 2; ++t) {      // N sub-frag
            int gn = j0 + wn * 32 + t * 16 + l16;
            #pragma unroll
            for (int rr = 0; rr < 8; ++rr) {
                int gm = i0 + wm * 32 + u * 16 + (half << 3) + rr;
                float val = cf[u][t].f[rr] * scale;
                if (HAS_BIAS) val += bias[gn];
                if (RELU)     val  = fmaxf(val, 0.0f);
                if (HAS_RES || OUT_F || OUT_B) {
                    long long cix = coff + (long long)gm * ldc + gn;
                    if (HAS_RES)  val += resid[cix];
                    if (OUT_F)    outF[cix] = val;
                    if (OUT_B)    outB[cix] = f2bf(val);
                }
                if (OUT_BT)   outB[(long long)gn * ldct + gm] = f2bf(val);
            }
        }
    }
}

// ---------------------------------------------------------------------------
// Host-side orchestration
// ---------------------------------------------------------------------------
static inline char* carve(char*& p, size_t bytes) {
    char* r = p;
    p += (bytes + 255) & ~(size_t)255;
    return r;
}

template <int HAS_BIAS, int RELU, int HAS_RES, int OUT_F, int OUT_B, int OUT_BT>
static void launch_gemm(hipStream_t st,
                        const unsigned short* A, const unsigned short* Bt,
                        const float* bias, const float* resid,
                        float* outF, unsigned short* outB,
                        int M, int N, int K, int lda, int ldb, int ldc, int ldct,
                        long long sA0, long long sA1, long long sB0, long long sB1,
                        long long sC0, long long sC1,
                        int batch, int inner, float scale) {
    dim3 g(N / 64, M / 128, batch);
    k_gemm_bf16_wmma<HAS_BIAS, RELU, HAS_RES, OUT_F, OUT_B, OUT_BT>
        <<<g, 256, 0, st>>>(A, Bt, bias, resid, outF, outB,
                            M, N, K, lda, ldb, ldc, ldct,
                            sA0, sA1, sB0, sB1, sC0, sC1, inner, scale);
}

// Transposing weight conversion: W (K x N f32) -> Wt (N x K bf16)
static void to_bf16_t(hipStream_t st, const float* W, unsigned short* Wt, int K, int N) {
    dim3 g(N / 32, K / 32);
    k_f32_to_bf16_t<<<g, 256, 0, st>>>(W, Wt, K, N);
}

extern "C" void kernel_launch(void* const* d_in, const int* in_sizes, int n_in,
                              void* d_out, int out_size, void* d_ws, size_t ws_size,
                              hipStream_t stream) {
    (void)in_sizes; (void)n_in; (void)out_size; (void)ws_size;

    const int*   tokens = (const int*)  d_in[0];
    const float* emb_W  = (const float*)d_in[1];
    const float* Wq     = (const float*)d_in[2];
    const float* bq     = (const float*)d_in[3];
    const float* Wk     = (const float*)d_in[4];
    const float* bk     = (const float*)d_in[5];
    const float* Wv     = (const float*)d_in[6];
    const float* bv     = (const float*)d_in[7];
    const float* Wo     = (const float*)d_in[8];
    const float* bo     = (const float*)d_in[9];
    const float* ln1_g  = (const float*)d_in[10];
    const float* ln1_b  = (const float*)d_in[11];
    const float* ln2_g  = (const float*)d_in[12];
    const float* ln2_b  = (const float*)d_in[13];
    const float* W1     = (const float*)d_in[14];
    const float* b1     = (const float*)d_in[15];
    const float* W2     = (const float*)d_in[16];
    const float* b2     = (const float*)d_in[17];
    const float* lnf_g  = (const float*)d_in[18];
    const float* lnf_b  = (const float*)d_in[19];
    const float* head_W = (const float*)d_in[20];
    const float* head_b = (const float*)d_in[21];
    float* out = (float*)d_out;

    const long long TD  = (long long)TLM_T * TLM_D;
    const long long TFF = (long long)TLM_T * TLM_FF;
    const long long SCN = (long long)TLM_B * TLM_H * TLM_S * TLM_S;

    char* wp = (char*)d_ws;
    float*          x      = (float*)         carve(wp, TD  * sizeof(float));
    unsigned short* lnb    = (unsigned short*)carve(wp, TD  * 2);
    unsigned short* qb     = (unsigned short*)carve(wp, TD  * 2);
    unsigned short* kb     = (unsigned short*)carve(wp, TD  * 2);
    unsigned short* vT     = (unsigned short*)carve(wp, TD  * 2);  // [D][T]
    unsigned short* attnb  = (unsigned short*)carve(wp, TD  * 2);
    unsigned short* hb     = (unsigned short*)carve(wp, TFF * 2);
    float*          scores = (float*)         carve(wp, SCN * sizeof(float));
    unsigned short* probs  = (unsigned short*)carve(wp, SCN * 2);
    unsigned short* wbuf   = (unsigned short*)carve(wp, (long long)TLM_D * TLM_V * 2);

    // ---- embedding + positional encoding
    k_embed_pe<<<TLM_T, 256, 0, stream>>>(tokens, emb_W, x);

    const long long DD  = (long long)TLM_D * TLM_D;
    const long long DFF = (long long)TLM_D * TLM_FF;
    const long long SD  = (long long)TLM_S * TLM_D;
    const long long SS  = (long long)TLM_S * TLM_S;

    for (int l = 0; l < TLM_L; ++l) {
        // ---- ln1 -> bf16
        k_layernorm_bf16<<<TLM_T, 256, 0, stream>>>(x, ln1_g + (long long)l * TLM_D,
                                                    ln1_b + (long long)l * TLM_D, lnb, TLM_D);
        // ---- Q/K projections (M=T, N=D, K=D): bias, bf16 out
        to_bf16_t(stream, Wq + l * DD, wbuf, TLM_D, TLM_D);
        launch_gemm<1,0,0,0,1,0>(stream, lnb, wbuf, bq + (long long)l * TLM_D, nullptr,
                                 nullptr, qb, TLM_T, TLM_D, TLM_D, TLM_D, TLM_D, TLM_D, 0,
                                 0, 0, 0, 0, 0, 0, 1, 1, 1.0f);
        to_bf16_t(stream, Wk + l * DD, wbuf, TLM_D, TLM_D);
        launch_gemm<1,0,0,0,1,0>(stream, lnb, wbuf, bk + (long long)l * TLM_D, nullptr,
                                 nullptr, kb, TLM_T, TLM_D, TLM_D, TLM_D, TLM_D, TLM_D, 0,
                                 0, 0, 0, 0, 0, 0, 1, 1, 1.0f);
        // ---- V projection: bf16 out TRANSPOSED -> vT[D][T]
        to_bf16_t(stream, Wv + l * DD, wbuf, TLM_D, TLM_D);
        launch_gemm<1,0,0,0,0,1>(stream, lnb, wbuf, bv + (long long)l * TLM_D, nullptr,
                                 nullptr, vT, TLM_T, TLM_D, TLM_D, TLM_D, TLM_D, TLM_D, TLM_T,
                                 0, 0, 0, 0, 0, 0, 1, 1, 1.0f);

        // ---- scores = (Q x K^T) / sqrt(HD), batched over (b,h): f32 out
        launch_gemm<0,0,0,1,0,0>(stream, qb, kb, nullptr, nullptr, scores, nullptr,
                                 TLM_S, TLM_S, TLM_HD, TLM_D, TLM_D, TLM_S, 0,
                                 SD, TLM_HD, SD, TLM_HD,
                                 (long long)TLM_H * SS, SS,
                                 TLM_B * TLM_H, TLM_H, 0.125f);

        // ---- causal softmax -> bf16 probs
        k_softmax_causal<<<TLM_B * TLM_H * TLM_S, 256, 0, stream>>>(scores, probs, TLM_S);

        // ---- attn_out = P x V = P x (vT slab)^T, batched over (b,h): bf16 out
        //      vT slab for (b,h): base = h*HD*T + b*S, row n stride = T
        launch_gemm<0,0,0,0,1,0>(stream, probs, vT, nullptr, nullptr, nullptr, attnb,
                                 TLM_S, TLM_HD, TLM_S, TLM_S, TLM_T, TLM_D, 0,
                                 (long long)TLM_H * SS, SS,
                                 (long long)TLM_S, (long long)TLM_HD * TLM_T,
                                 SD, TLM_HD,
                                 TLM_B * TLM_H, TLM_H, 1.0f);

        // ---- output projection + residual -> x  (bias, resid, f32 out)
        to_bf16_t(stream, Wo + l * DD, wbuf, TLM_D, TLM_D);
        launch_gemm<1,0,1,1,0,0>(stream, attnb, wbuf, bo + (long long)l * TLM_D, x,
                                 x, nullptr, TLM_T, TLM_D, TLM_D, TLM_D, TLM_D, TLM_D, 0,
                                 0, 0, 0, 0, 0, 0, 1, 1, 1.0f);

        // ---- ln2 -> bf16
        k_layernorm_bf16<<<TLM_T, 256, 0, stream>>>(x, ln2_g + (long long)l * TLM_D,
                                                    ln2_b + (long long)l * TLM_D, lnb, TLM_D);
        // ---- FFN: h = relu(lnb x W1 + b1) ; x += h x W2 + b2
        to_bf16_t(stream, W1 + l * DFF, wbuf, TLM_D, TLM_FF);
        launch_gemm<1,1,0,0,1,0>(stream, lnb, wbuf, b1 + (long long)l * TLM_FF, nullptr,
                                 nullptr, hb, TLM_T, TLM_FF, TLM_D, TLM_D, TLM_D, TLM_FF, 0,
                                 0, 0, 0, 0, 0, 0, 1, 1, 1.0f);
        to_bf16_t(stream, W2 + l * DFF, wbuf, TLM_FF, TLM_D);
        launch_gemm<1,0,1,1,0,0>(stream, hb, wbuf, b2 + (long long)l * TLM_D, x,
                                 x, nullptr, TLM_T, TLM_D, TLM_FF, TLM_FF, TLM_FF, TLM_D, 0,
                                 0, 0, 0, 0, 0, 0, 1, 1, 1.0f);
    }

    // ---- final LN + vocab head
    k_layernorm_bf16<<<TLM_T, 256, 0, stream>>>(x, lnf_g, lnf_b, lnb, TLM_D);
    to_bf16_t(stream, head_W, wbuf, TLM_D, TLM_V);
    launch_gemm<1,0,0,1,0,0>(stream, lnb, wbuf, head_b, nullptr, out, nullptr,
                             TLM_T, TLM_V, TLM_D, TLM_D, TLM_D, TLM_V, 0,
                             0, 0, 0, 0, 0, 0, 1, 1, 1.0f);
}
